// SimpleLSTM_2705829397165
// MI455X (gfx1250) — matrix operand order repaired and checked by reading, not compiled, not run on hardware
//
#include <hip/hip_runtime.h>
#include <hip/hip_bf16.h>

// ---------------------------------------------------------------------------
// LSTM (B=64, T=512, I=512, H=1024, O=512) for gfx1250 (MI455X).
// bf16 WMMA (v_wmma_f32_16x16x32_bf16) everywhere, f32 accumulate.
//   K1: f32 -> bf16 converts (x, W_ih, W_hh, W_fc)
//   K2: xg[t][b][4H] = x @ W_ih^T + b_ih + b_hh   (software-pipelined GEMM)
//   K3: persistent scan kernel, 64 WGs, grid barrier per timestep.
//       Waves split K (not M): 4 WMMAs share one B fragment per k-step,
//       K-halves reduced in LDS during the elementwise phase.
//   K4: out = h_T @ W_fc^T + b_fc
// Software pipelining is enforced with sched_group_barrier (verified in asm:
// clause of loads followed by WMMAs with partial s_wait_loadcnt).
// ---------------------------------------------------------------------------

#define B_ 64
#define T_ 512
#define I_ 512
#define H_ 1024
#define G4_ 4096   // 4*H
#define O_ 512
#define NWG_SCAN 64

typedef __attribute__((ext_vector_type(16))) __bf16 v16bf;
typedef __attribute__((ext_vector_type(8)))  float  v8f;

// scheduling groups: 0x020 = VMEM read, 0x008 = MFMA/WMMA
#if defined(__has_builtin)
#if __has_builtin(__builtin_amdgcn_sched_group_barrier)
#define SCHED_VMEM_THEN_WMMA(nld, nmma)                        \
    do {                                                       \
        __builtin_amdgcn_sched_group_barrier(0x020, (nld), 0); \
        __builtin_amdgcn_sched_group_barrier(0x008, (nmma), 0);\
    } while (0)
#endif
#endif
#ifndef SCHED_VMEM_THEN_WMMA
#define SCHED_VMEM_THEN_WMMA(nld, nmma) do {} while (0)
#endif

// ---- WMMA fragment loaders (layouts per CDNA5 ISA §7.12.2) -----

// A-matrix 16x32 bf16 fragment from row-major [rows][ldk] memory.
// lane<16: m=row0+lane,      K = {k0..k0+7, k0+16..k0+23}
// lane>=16: m=row0+lane-16,  K = {k0+8..k0+15, k0+24..k0+31}
__device__ __forceinline__ v16bf load_frag_a(const __bf16* base, int row0,
                                             int k0, int ldk, int lane) {
    int m  = row0 + (lane & 15);
    int kb = k0 + ((lane >> 4) << 3);
    const __bf16* p = base + (size_t)m * ldk + kb;
    union { uint4 u[2]; v16bf v; } r;
    r.u[0] = *(const uint4*)(p);        // K = kb .. kb+7
    r.u[1] = *(const uint4*)(p + 16);   // K = kb+16 .. kb+23
    return r.v;
}

// B-matrix 32x16 bf16 fragment. Memory holds W[n][k] row-major (K contiguous),
// i.e. B[k][n] = W[n][k]. lane<16: n=col0+lane, K = {k0..k0+7, k0+16..k0+23};
// lane>=16: K = {k0+8..k0+15, k0+24..k0+31}.
__device__ __forceinline__ v16bf load_frag_b(const __bf16* base, int col0,
                                             int k0, int ldk, int lane) {
    int n  = col0 + (lane & 15);
    int kb = k0 + ((lane >> 4) << 3);
    const __bf16* p = base + (size_t)n * ldk + kb;
    union { uint4 u[2]; v16bf v; } r;
    r.u[0] = *(const uint4*)(p);        // K = kb .. kb+7
    r.u[1] = *(const uint4*)(p + 16);   // K = kb+16 .. kb+23
    return r.v;
}

__device__ __forceinline__ v8f wmma_bf16(v16bf a, v16bf b, v8f c) {
    return __builtin_amdgcn_wmma_f32_16x16x32_bf16(
        /*neg_a=*/false, a, /*neg_b=*/false, b,
        /*c_mod=*/(short)0, c, /*reuse_a=*/false, /*reuse_b=*/false);
}

__device__ __forceinline__ float sigf(float x) {
    return 1.0f / (1.0f + __expf(-x));
}

// ---- K1: f32 -> bf16 convert ----------------------------------------------
__global__ void cvt_bf16_kernel(const float* __restrict__ s,
                                __bf16* __restrict__ d, int n) {
    int i = blockIdx.x * blockDim.x + threadIdx.x;
    int stride = gridDim.x * blockDim.x;
    for (; i < n; i += stride) d[i] = (__bf16)s[i];
}

// ---- K2: xg GEMM (software-pipelined) -------------------------------------
// A = x_bf16 [B*T][I] (row bt = b*T + t), Bmat = W_ih_bf16 [4H][I].
// Output transposed to xg[t][b][4H] (f32) so the scan reads contiguously.
// Block: 256 thr (8 waves) covering M=32 x N=256. Wave: 16(M) x 64(N).
__global__ __launch_bounds__(256)
void xg_gemm_kernel(const __bf16* __restrict__ x, const __bf16* __restrict__ Wih,
                    const float* __restrict__ bih, const float* __restrict__ bhh,
                    float* __restrict__ xg) {
    const int lane = threadIdx.x & 31;
    const int wv   = threadIdx.x >> 5;      // 0..7
    const int m0   = blockIdx.y * 32 + (wv & 1) * 16;
    const int n0   = blockIdx.x * 256 + (wv >> 1) * 64;

    v8f acc[4] = {v8f{}, v8f{}, v8f{}, v8f{}};

    // prologue: fragments for k0 = 0
    v16bf a = load_frag_a(x, m0, 0, I_, lane);
    v16bf b[4];
#pragma unroll
    for (int nt = 0; nt < 4; ++nt) b[nt] = load_frag_b(Wih, n0 + nt * 16, 0, I_, lane);

#pragma unroll
    for (int k0 = 0; k0 < I_ - 32; k0 += 32) {
        // prefetch next k-step while the matrix pipe chews on this one
        v16bf an = load_frag_a(x, m0, k0 + 32, I_, lane);
        v16bf bn[4];
#pragma unroll
        for (int nt = 0; nt < 4; ++nt)
            bn[nt] = load_frag_b(Wih, n0 + nt * 16, k0 + 32, I_, lane);
#pragma unroll
        for (int nt = 0; nt < 4; ++nt) acc[nt] = wmma_bf16(a, b[nt], acc[nt]);
        a = an;
#pragma unroll
        for (int nt = 0; nt < 4; ++nt) b[nt] = bn[nt];
        SCHED_VMEM_THEN_WMMA(10, 4);
    }
#pragma unroll
    for (int nt = 0; nt < 4; ++nt) acc[nt] = wmma_bf16(a, b[nt], acc[nt]);

#pragma unroll
    for (int nt = 0; nt < 4; ++nt) {
        int n = n0 + nt * 16 + (lane & 15);
        float bias = bih[n] + bhh[n];
#pragma unroll
        for (int r = 0; r < 8; ++r) {
            int m = m0 + r + ((lane >> 4) << 3);   // m = b*T + t
            int bb = m >> 9;          // /T_
            int tt = m & (T_ - 1);
            xg[((size_t)(tt * B_ + bb)) * G4_ + n] = acc[nt][r] + bias;
        }
    }
}

// ---- K3: persistent LSTM scan ---------------------------------------------
// 64 WGs x 256 threads. WG w owns h-columns [16w, 16w+16). Per step the WG
// computes all 4 gates for its 16 columns: GEMM h[64x1024] @ W_hh^T slice.
// Wave wv: gate = wv>>1, K-half = wv&1 (512 of K=1024). Each wave computes
// all four 16-row M-tiles over its K-half: one B fragment feeds 4 WMMAs.
// K-half partials are summed in LDS in the elementwise phase (with xg).
// c lives in LDS (never shared); h (bf16, 128 KB) double-buffered in global,
// exchanged via a monotonic atomic grid barrier each step.
__global__ __launch_bounds__(256, 1)
void lstm_scan_kernel(const float* __restrict__ xg, const __bf16* __restrict__ Whh,
                      __bf16* __restrict__ h0, __bf16* __restrict__ h1,
                      unsigned* __restrict__ barrier) {
    __shared__ float gates_s[2][4][B_][16];   // [khalf][gate][row][col]
    __shared__ float c_s[B_][16];

    const int tid   = threadIdx.x;
    const int lane  = tid & 31;
    const int wv    = tid >> 5;
    const int gw    = wv >> 1;                 // gate 0..3 (i,f,g,o)
    const int kh    = wv & 1;                  // K-half
    const int kbase = kh * (H_ / 2);           // 0 or 512
    const int hcol0 = blockIdx.x * 16;         // owned h columns
    const int nglob = gw * H_ + hcol0;         // column base in [0,4H)

    for (int e = tid; e < B_ * 16; e += 256) c_s[e >> 4][e & 15] = 0.0f;
    __syncthreads();

    __bf16* hbuf[2] = {h0, h1};

    for (int t = 0; t < T_; ++t) {
        const __bf16* hcur = hbuf[t & 1];
        __bf16* hnext = hbuf[(t + 1) & 1];

        // warm L2 with next step's xg slab (read-once from HBM):
        // 4 gates x 64 rows of 16 f32 = 256 64B chunks -> one per thread.
        if (t + 1 < T_) {
            const float* xgn = xg + (size_t)(t + 1) * B_ * G4_
                             + (size_t)(tid & 63) * G4_ + (tid >> 6) * H_ + hcol0;
            __builtin_prefetch(xgn, 0, 1);
        }

        v8f acc[4] = {v8f{}, v8f{}, v8f{}, v8f{}};

        // prologue fragments (k0 = kbase)
        v16bf b = load_frag_b(Whh, nglob, kbase, H_, lane);
        v16bf a[4];
#pragma unroll
        for (int mt = 0; mt < 4; ++mt)
            a[mt] = load_frag_a(hcur, mt * 16, kbase, H_, lane);

#pragma unroll
        for (int i = 0; i < (H_ / 2) / 32 - 1; ++i) {
            const int k0 = kbase + i * 32;
            v16bf bn = load_frag_b(Whh, nglob, k0 + 32, H_, lane);
            v16bf an[4];
#pragma unroll
            for (int mt = 0; mt < 4; ++mt)
                an[mt] = load_frag_a(hcur, mt * 16, k0 + 32, H_, lane);
#pragma unroll
            for (int mt = 0; mt < 4; ++mt) acc[mt] = wmma_bf16(a[mt], b, acc[mt]);
            b = bn;
#pragma unroll
            for (int mt = 0; mt < 4; ++mt) a[mt] = an[mt];
            SCHED_VMEM_THEN_WMMA(10, 4);
        }
#pragma unroll
        for (int mt = 0; mt < 4; ++mt) acc[mt] = wmma_bf16(a[mt], b, acc[mt]);

        // stage K-half partial sums to LDS
        const int ncol = lane & 15;
#pragma unroll
        for (int mt = 0; mt < 4; ++mt)
#pragma unroll
            for (int r = 0; r < 8; ++r) {
                int m = mt * 16 + r + ((lane >> 4) << 3);
                gates_s[kh][gw][m][ncol] = acc[mt][r];
            }
        __syncthreads();

        // elementwise LSTM cell update (reduce K-halves + xg here)
        const float* xgt = xg + (size_t)t * B_ * G4_;
        for (int e = tid; e < B_ * 16; e += 256) {
            int m = e >> 4, col = e & 15;
            const float* xr = xgt + (size_t)m * G4_ + hcol0 + col;
            float ig = sigf (gates_s[0][0][m][col] + gates_s[1][0][m][col] + xr[0 * H_]);
            float fg = sigf (gates_s[0][1][m][col] + gates_s[1][1][m][col] + xr[1 * H_]);
            float gg = tanhf(gates_s[0][2][m][col] + gates_s[1][2][m][col] + xr[2 * H_]);
            float og = sigf (gates_s[0][3][m][col] + gates_s[1][3][m][col] + xr[3 * H_]);
            float c  = fg * c_s[m][col] + ig * gg;
            c_s[m][col] = c;
            hnext[(size_t)m * H_ + hcol0 + col] = (__bf16)(og * tanhf(c));
        }

        // ---- grid-wide barrier (monotonic counter) ----
        __builtin_amdgcn_fence(__ATOMIC_RELEASE, "agent");  // flush h stores
        __syncthreads();
        if (tid == 0) {
            __hip_atomic_fetch_add(barrier, 1u, __ATOMIC_RELAXED,
                                   __HIP_MEMORY_SCOPE_AGENT);
            unsigned target = (unsigned)(t + 1) * gridDim.x;
            while (__hip_atomic_load(barrier, __ATOMIC_RELAXED,
                                     __HIP_MEMORY_SCOPE_AGENT) < target)
                __builtin_amdgcn_s_sleep(1);
        }
        __syncthreads();
        __builtin_amdgcn_fence(__ATOMIC_ACQUIRE, "agent");  // invalidate for h reads
    }
}

// ---- K4: final FC ---------------------------------------------------------
// out[64][512] = h_T(bf16) @ W_fc^T + b_fc. 128 waves, one 16x16 tile each.
__global__ __launch_bounds__(256)
void fc_kernel(const __bf16* __restrict__ h, const __bf16* __restrict__ Wfc,
               const float* __restrict__ bfc, float* __restrict__ out) {
    const int lane = threadIdx.x & 31;
    const int w    = blockIdx.x * 8 + (threadIdx.x >> 5);  // 0..127
    const int m0   = (w & 3) * 16;
    const int n0   = (w >> 2) * 16;

    v8f acc = {};
    v16bf a = load_frag_a(h,   m0, 0, H_, lane);
    v16bf b = load_frag_b(Wfc, n0, 0, H_, lane);
#pragma unroll 4
    for (int k0 = 0; k0 < H_ - 32; k0 += 32) {
        v16bf an = load_frag_a(h,   m0, k0 + 32, H_, lane);
        v16bf bn = load_frag_b(Wfc, n0, k0 + 32, H_, lane);
        acc = wmma_bf16(a, b, acc);
        a = an; b = bn;
        SCHED_VMEM_THEN_WMMA(4, 1);
    }
    acc = wmma_bf16(a, b, acc);

    int n = n0 + (lane & 15);
    float bias = bfc[n];
#pragma unroll
    for (int r = 0; r < 8; ++r) {
        int m = m0 + r + ((lane >> 4) << 3);
        out[(size_t)m * O_ + n] = acc[r] + bias;
    }
}

// ---------------------------------------------------------------------------
extern "C" void kernel_launch(void* const* d_in, const int* in_sizes, int n_in,
                              void* d_out, int out_size, void* d_ws, size_t ws_size,
                              hipStream_t stream) {
    (void)in_sizes; (void)n_in; (void)out_size; (void)ws_size;

    const float* x    = (const float*)d_in[0];   // [64][512][512]
    const float* Wih  = (const float*)d_in[1];   // [4096][512]
    const float* Whh  = (const float*)d_in[2];   // [4096][1024]
    const float* bih  = (const float*)d_in[3];   // [4096]
    const float* bhh  = (const float*)d_in[4];   // [4096]
    const float* Wfc  = (const float*)d_in[5];   // [512][1024]
    const float* bfc  = (const float*)d_in[6];   // [512]
    float* out = (float*)d_out;                  // [64][512]

    // workspace carve-up (256-B aligned)
    char* ws = (char*)d_ws;
    size_t off = 0;
    auto carve = [&](size_t bytes) -> void* {
        void* p = ws + off;
        off = (off + bytes + 255) & ~(size_t)255;
        return p;
    };
    __bf16* x_bf   = (__bf16*)carve((size_t)B_ * T_ * I_ * 2);   // 33.5 MB
    __bf16* Wih_bf = (__bf16*)carve((size_t)G4_ * I_ * 2);       // 4 MB
    __bf16* Whh_bf = (__bf16*)carve((size_t)G4_ * H_ * 2);       // 8 MB
    __bf16* Wfc_bf = (__bf16*)carve((size_t)O_ * H_ * 2);        // 1 MB
    __bf16* h0     = (__bf16*)carve((size_t)B_ * H_ * 2);        // 128 KB
    __bf16* h1     = (__bf16*)carve((size_t)B_ * H_ * 2);        // 128 KB
    unsigned* bar  = (unsigned*)carve(256);
    float* xg      = (float*)carve((size_t)T_ * B_ * G4_ * 4);   // 537 MB

    // per-launch init (capture-safe, deterministic)
    hipMemsetAsync(bar, 0, 256, stream);
    hipMemsetAsync(h0, 0, (size_t)B_ * H_ * 2, stream);

    // K1: converts
    cvt_bf16_kernel<<<4096, 256, 0, stream>>>(x,   x_bf,   B_ * T_ * I_);
    cvt_bf16_kernel<<<2048, 256, 0, stream>>>(Wih, Wih_bf, G4_ * I_);
    cvt_bf16_kernel<<<4096, 256, 0, stream>>>(Whh, Whh_bf, G4_ * H_);
    cvt_bf16_kernel<<<512,  256, 0, stream>>>(Wfc, Wfc_bf, O_ * H_);

    // K2: xg = x @ W_ih^T + b_ih + b_hh   (stored [T][B][4H])
    xg_gemm_kernel<<<dim3(G4_ / 256, (B_ * T_) / 32), 256, 0, stream>>>(
        x_bf, Wih_bf, bih, bhh, xg);

    // K3: recurrent scan (persistent, grid barrier per step)
    lstm_scan_kernel<<<NWG_SCAN, 256, 0, stream>>>(xg, Whh_bf, h0, h1, bar);

    // final h is in h0 (T even). K4: FC.
    fc_kernel<<<16, 256, 0, stream>>>(h0, Wfc_bf, bfc, out);
}
